// PointPillarsSem_9363028706389
// MI455X (gfx1250) — matrix operand order repaired and checked by reading, not compiled, not run on hardware
//
#include <hip/hip_runtime.h>
#include <hip/hip_bf16.h>

typedef __attribute__((ext_vector_type(16))) _Float16 v16h;
typedef __attribute__((ext_vector_type(8)))  _Float16 v8h;
typedef __attribute__((ext_vector_type(8)))  float    v8f;

#define XN 512
#define YN 512
#define BATCH 4
#define GRID_NUM (BATCH * XN * YN)   // 1,048,576

// ---------------------------------------------------------------------------
// Pack conv weights [4][32cout][32cin][3][3] f32 into per-lane WMMA A-fragments
// wf layout: [layer][tap(9)][half(2)][lane(32)][16] f16
// A 16x32 f16 fragment: lane l -> M=l&15 ; elem e -> K = e + (l<16?0:16)
// ---------------------------------------------------------------------------
__global__ void k_wfrag(const float* __restrict__ w2d, _Float16* __restrict__ wf) {
    int t = blockIdx.x * blockDim.x + threadIdx.x;      // 4*9*2*32 = 2304
    if (t >= 4 * 9 * 2 * 32) return;
    int lane  = t & 31;
    int half  = (t >> 5) & 1;
    int tap   = (t >> 6) % 9;
    int layer = t / (9 * 2 * 32);
    int ky = tap / 3, kx = tap % 3;
    int m   = (lane & 15) + 16 * half;                  // cout
    int klo = (lane < 16) ? 0 : 16;
    _Float16* dst = wf + (size_t)t * 16;
    for (int e = 0; e < 16; ++e) {
        int k = e + klo;                                // cin
        dst[e] = (_Float16)w2d[((((size_t)layer * 32 + m) * 32 + k) * 3 + ky) * 3 + kx];
    }
}

// ---------------------------------------------------------------------------
// Stage 1: per-point MLPs through vfe0's pointwise conv; also emit voxel ids.
// ---------------------------------------------------------------------------
__global__ void k_stage1(const int* __restrict__ bids, const float* __restrict__ pts,
                         const float* w0, const float* b0, const float* s0, const float* t0,
                         const float* w1, const float* b1, const float* s1, const float* t1,
                         const float* wv0, const float* bv0, const float* sv0, const float* tv0,
                         int N, int* __restrict__ gids, float* __restrict__ oA) {
    int i = blockIdx.x * blockDim.x + threadIdx.x;
    if (i >= N) return;
    float p[6];
    for (int c = 0; c < 6; ++c) p[c] = pts[(size_t)c * N + i];
    int xg = (int)floorf(p[0] * (float)XN);
    int yg = (int)floorf(p[1] * (float)YN);
    gids[i] = bids[i] * (XN * YN) + yg * XN + xg;

    float h0[16];
    for (int o = 0; o < 16; ++o) {
        float a = b0[o];
        for (int c = 0; c < 6; ++c) a += w0[o * 6 + c] * p[c];
        h0[o] = fmaxf(a * s0[o] + t0[o], 0.f);
    }
    float h1[16];
    for (int o = 0; o < 16; ++o) {
        float a = b1[o];
        for (int c = 0; c < 16; ++c) a += w1[o * 16 + c] * h0[c];
        h1[o] = fmaxf(a * s1[o] + t1[o], 0.f);
    }
    float dx = p[0] - ((float)xg + 0.5f) * (1.0f / XN);
    float dy = p[1] - ((float)yg + 0.5f) * (1.0f / YN);
    float* out = oA + (size_t)i * 16;
    for (int o = 0; o < 16; ++o) {
        float a = bv0[o];
        for (int c = 0; c < 16; ++c) a += wv0[o * 18 + c] * h1[c];
        a += wv0[o * 18 + 16] * dx + wv0[o * 18 + 17] * dy;
        out[o] = fmaxf(a * sv0[o] + tv0[o], 0.f);
    }
}

// ---------------------------------------------------------------------------
__global__ void k_zero4(float* __restrict__ p, size_t n) {
    size_t i = ((size_t)blockIdx.x * blockDim.x + threadIdx.x) * 4;
    if (i + 3 < n) {
        *(float4*)(p + i) = make_float4(0.f, 0.f, 0.f, 0.f);
    } else {
        for (size_t j = i; j < n; ++j) p[j] = 0.f;
    }
}

// Scatter-max: values are post-ReLU >= 0, so float max == u32 max on bits.
__global__ void k_scat(const float* __restrict__ vals, const int* __restrict__ gids,
                       unsigned* __restrict__ grid, int N, int sh) {
    long long t = (long long)blockIdx.x * blockDim.x + threadIdx.x;
    if (t >= ((long long)N << sh)) return;
    int c = (int)(t & ((1 << sh) - 1));
    int i = (int)(t >> sh);
    unsigned v = __float_as_uint(vals[((size_t)i << sh) + c]);
    atomicMax(grid + ((size_t)gids[i] << sh) + c, v);
}

// concat(o, grid[gid]) -> pointwise conv 32->OUTC + affine + ReLU
template <int OUTC>
__global__ void k_vfe(const float* __restrict__ oin, const float* __restrict__ gridS,
                      const int* __restrict__ gids, const float* w, const float* bb,
                      const float* ss, const float* tt, int N, float* __restrict__ oout) {
    int i = blockIdx.x * blockDim.x + threadIdx.x;
    if (i >= N) return;
    float h[32];
    const float* oi = oin + (size_t)i * 16;
    const float* g  = gridS + (size_t)gids[i] * 16;
    for (int c = 0; c < 16; ++c) { h[c] = oi[c]; h[16 + c] = g[c]; }
    float* out = oout + (size_t)i * OUTC;
    for (int o = 0; o < OUTC; ++o) {
        float a = bb[o];
        for (int c = 0; c < 32; ++c) a += w[o * 32 + c] * h[c];
        out[o] = fmaxf(a * ss[o] + tt[o], 0.f);
    }
}

__global__ void k_cvt(const float* __restrict__ in, _Float16* __restrict__ out, size_t n) {
    size_t i = ((size_t)blockIdx.x * blockDim.x + threadIdx.x) * 4;
    if (i + 3 < n) {
        float4 v = *(const float4*)(in + i);
        out[i + 0] = (_Float16)v.x; out[i + 1] = (_Float16)v.y;
        out[i + 2] = (_Float16)v.z; out[i + 3] = (_Float16)v.w;
    } else {
        for (size_t j = i; j < n; ++j) out[j] = (_Float16)in[j];
    }
}

// ---------------------------------------------------------------------------
// Dilated 3x3 conv (32->32 ch) as implicit GEMM on WMMA.
// NHWC f16, C=32. One wave -> 2 adjacent 16-pixel tiles x 32 couts (same y),
// sharing A-fragments: each A load feeds 2 WMMAs. 36 WMMAs per wave.
// Fused (x+b)*s+t ReLU epilogue, packed 16B f16 stores.
// ---------------------------------------------------------------------------
__global__ void __launch_bounds__(256) k_conv(const _Float16* __restrict__ in,
                                              _Float16* __restrict__ out,
                                              const _Float16* __restrict__ wf,
                                              const float* __restrict__ bb,
                                              const float* __restrict__ ss,
                                              const float* __restrict__ tt, int dil) {
    int lane = threadIdx.x & 31;
    int wave = threadIdx.x >> 5;
    int st   = blockIdx.x * 8 + wave;        // supertiles: 4*512*(512/32) = 32768
    int tx0  = (st & 15) * 32;               // 32-pixel strip, two 16-pixel tiles
    int y    = (st >> 4) & 511;
    int bimg = st >> 13;

    int nlo  = lane & 15;                    // B-matrix column == pixel in tile
    int cinb = (lane < 16) ? 0 : 16;         // B-matrix K base for this lane half
    const _Float16* ibase = in + (size_t)bimg * XN * YN * 32;

    v8f acc00 = {}, acc01 = {};              // tile0: couts lo / hi
    v8f acc10 = {}, acc11 = {};              // tile1

#pragma unroll
    for (int ky = 0; ky < 3; ++ky) {
        int yy = y + (ky - 1) * dil;
        bool rowok = (yy >= 0) && (yy < YN);
        // prefetch next tap-row's activation lines while this row's WMMAs issue
        int yn = y + ky * dil;               // row used at ky+1
        if (ky < 2 && yn >= 0 && yn < YN)
            __builtin_prefetch((const void*)(ibase + ((size_t)(yn * XN + tx0 + nlo) * 32)), 0, 0);
#pragma unroll
        for (int kx = 0; kx < 3; ++kx) {
            int tap = ky * 3 + kx;
            const _Float16* wfp = wf + ((size_t)(tap * 2) * 32 + lane) * 16;
            v16h a0 = *(const v16h*)(wfp);
            v16h a1 = *(const v16h*)(wfp + 32 * 16);

            int xx0 = tx0 + nlo + (kx - 1) * dil;
            int xx1 = xx0 + 16;
            v16h bf0 = {}, bf1 = {};
            if (rowok && xx0 >= 0 && xx0 < XN)
                bf0 = *(const v16h*)(ibase + ((size_t)(yy * XN + xx0) * 32 + cinb));
            if (rowok && xx1 >= 0 && xx1 < XN)
                bf1 = *(const v16h*)(ibase + ((size_t)(yy * XN + xx1) * 32 + cinb));

            acc00 = __builtin_amdgcn_wmma_f32_16x16x32_f16(false, a0, false, bf0,
                                                           (short)0, acc00, false, false);
            acc01 = __builtin_amdgcn_wmma_f32_16x16x32_f16(false, a1, false, bf0,
                                                           (short)0, acc01, false, false);
            acc10 = __builtin_amdgcn_wmma_f32_16x16x32_f16(false, a0, false, bf1,
                                                           (short)0, acc10, false, false);
            acc11 = __builtin_amdgcn_wmma_f32_16x16x32_f16(false, a1, false, bf1,
                                                           (short)0, acc11, false, false);
        }
    }

    // D layout: VGPR r, lane l -> cout = r + (l<16?0:8) (+16 for *1), pixel = tx+(l&15)
    int cb = (lane < 16) ? 0 : 8;
    float sc[8], tc[8], bc[8], sc1[8], tc1[8], bc1[8];
#pragma unroll
    for (int r = 0; r < 8; ++r) {
        bc[r]  = bb[cb + r];      sc[r]  = ss[cb + r];      tc[r]  = tt[cb + r];
        bc1[r] = bb[cb + r + 16]; sc1[r] = ss[cb + r + 16]; tc1[r] = tt[cb + r + 16];
    }
#pragma unroll
    for (int t = 0; t < 2; ++t) {
        int px = tx0 + t * 16 + nlo;
        _Float16* obase = out + ((size_t)bimg * XN * YN + (size_t)y * XN + px) * 32;
        v8f a0 = t ? acc10 : acc00;
        v8f a1 = t ? acc11 : acc01;
        v8h r0, r1;
#pragma unroll
        for (int r = 0; r < 8; ++r) {
            r0[r] = (_Float16)fmaxf((a0[r] + bc[r])  * sc[r]  + tc[r],  0.f);
            r1[r] = (_Float16)fmaxf((a1[r] + bc1[r]) * sc1[r] + tc1[r], 0.f);
        }
        *(v8h*)(obase + cb)      = r0;
        *(v8h*)(obase + 16 + cb) = r1;
    }
}

// ---------------------------------------------------------------------------
// Bilinear gather from final f16 grid + concat(point0,point1) + w3 matmul.
// ---------------------------------------------------------------------------
__global__ void k_final(const int* __restrict__ bids, const float* __restrict__ pts,
                        const float* __restrict__ point0, const _Float16* __restrict__ img,
                        const float* __restrict__ w3, const float* __restrict__ b3,
                        float* __restrict__ out, int N) {
    int i = blockIdx.x * blockDim.x + threadIdx.x;
    if (i >= N) return;
    float ix = pts[i] * (float)XN - 1.0f;
    float iy = pts[(size_t)N + i] * (float)YN - 1.0f + (float)bids[i] * (float)YN;
    float x0 = floorf(ix), y0 = floorf(iy);
    float wx = ix - x0, wy = iy - y0;
    int x0i = (int)x0, y0i = (int)y0;
    const int H = BATCH * YN, W = XN;

    float p1[32];
    for (int c = 0; c < 32; ++c) p1[c] = 0.f;
#pragma unroll
    for (int t = 0; t < 4; ++t) {
        int yi = y0i + (t >> 1);
        int xi = x0i + (t & 1);
        float wgt = ((t & 1) ? wx : 1.f - wx) * ((t >> 1) ? wy : 1.f - wy);
        if (yi >= 0 && yi < H && xi >= 0 && xi < W) {
            const _Float16* q = img + ((size_t)yi * W + xi) * 32;
            v16h q0 = *(const v16h*)(q);
            v16h q1 = *(const v16h*)(q + 16);
#pragma unroll
            for (int c = 0; c < 16; ++c) {
                p1[c]      += wgt * (float)q0[c];
                p1[16 + c] += wgt * (float)q1[c];
            }
        }
    }
    const float* p0 = point0 + (size_t)i * 32;
    for (int k = 0; k < 20; ++k) {
        float a = b3[k];
        const float* wr = w3 + k * 64;
        for (int c = 0; c < 32; ++c) a += wr[c] * p0[c];
        for (int c = 0; c < 32; ++c) a += wr[32 + c] * p1[c];
        out[(size_t)k * N + i] = a;
    }
}

// ---------------------------------------------------------------------------
extern "C" void kernel_launch(void* const* d_in, const int* in_sizes, int n_in,
                              void* d_out, int out_size, void* d_ws, size_t ws_size,
                              hipStream_t stream) {
    const int N = in_sizes[0];
    const int*   bids = (const int*)d_in[0];
    const float* pts  = (const float*)d_in[1];
    const float *w0 = (const float*)d_in[3],  *b0 = (const float*)d_in[4],
                *s0 = (const float*)d_in[5],  *t0 = (const float*)d_in[6];
    const float *w1 = (const float*)d_in[7],  *b1 = (const float*)d_in[8],
                *s1 = (const float*)d_in[9],  *t1 = (const float*)d_in[10];
    const float *wv0 = (const float*)d_in[11], *bv0 = (const float*)d_in[12],
                *sv0 = (const float*)d_in[13], *tv0 = (const float*)d_in[14];
    const float *wv1 = (const float*)d_in[15], *bv1 = (const float*)d_in[16],
                *sv1 = (const float*)d_in[17], *tv1 = (const float*)d_in[18];
    const float *w2 = (const float*)d_in[19], *b2 = (const float*)d_in[20],
                *s2 = (const float*)d_in[21], *t2 = (const float*)d_in[22];
    const float *w2d = (const float*)d_in[23], *b2d = (const float*)d_in[24],
                *s2d = (const float*)d_in[25], *t2d = (const float*)d_in[26];
    const float *w3 = (const float*)d_in[27], *b3 = (const float*)d_in[28];
    float* out = (float*)d_out;

    char* ws = (char*)d_ws;
    size_t off = 0;
    auto alloc = [&](size_t bytes) -> char* {
        char* p = ws + off;
        off = (off + bytes + 255) & ~(size_t)255;
        return p;
    };
    int*       gids   = (int*)alloc((size_t)N * 4);
    float*     oA     = (float*)alloc((size_t)N * 16 * 4);
    float*     oB     = (float*)alloc((size_t)N * 16 * 4);
    float*     point0 = (float*)alloc((size_t)N * 32 * 4);
    float*     gridS  = (float*)alloc((size_t)GRID_NUM * 16 * 4);
    float*     gridF  = (float*)alloc((size_t)GRID_NUM * 32 * 4);
    _Float16*  gA     = (_Float16*)alloc((size_t)GRID_NUM * 32 * 2);
    _Float16*  gB     = (_Float16*)alloc((size_t)GRID_NUM * 32 * 2);
    _Float16*  wfrag  = (_Float16*)alloc((size_t)4 * 9 * 2 * 32 * 16 * 2);

    const int TB = 256;
    const size_t g16 = (size_t)GRID_NUM * 16;
    const size_t g32 = (size_t)GRID_NUM * 32;

    // weight fragments for the 4 conv layers
    k_wfrag<<<9, TB, 0, stream>>>(w2d, wfrag);

    // stage 1: point MLPs + vfe0 pconv + voxel ids
    k_stage1<<<(N + TB - 1) / TB, TB, 0, stream>>>(bids, pts, w0, b0, s0, t0,
                                                   w1, b1, s1, t1, wv0, bv0, sv0, tv0,
                                                   N, gids, oA);

    // VFE block 1: scatter-max(oA) -> gather -> pconv wv1 -> oB
    k_zero4<<<(unsigned)((g16 / 4 + TB - 1) / TB), TB, 0, stream>>>(gridS, g16);
    k_scat<<<(unsigned)(((size_t)N * 16 + TB - 1) / TB), TB, 0, stream>>>(oA, gids, (unsigned*)gridS, N, 4);
    k_vfe<16><<<(N + TB - 1) / TB, TB, 0, stream>>>(oA, gridS, gids, wv1, bv1, sv1, tv1, N, oB);

    // VFE block 2: scatter-max(oB) -> gather -> pconv w2 -> point0
    k_zero4<<<(unsigned)((g16 / 4 + TB - 1) / TB), TB, 0, stream>>>(gridS, g16);
    k_scat<<<(unsigned)(((size_t)N * 16 + TB - 1) / TB), TB, 0, stream>>>(oB, gids, (unsigned*)gridS, N, 4);
    k_vfe<32><<<(N + TB - 1) / TB, TB, 0, stream>>>(oB, gridS, gids, w2, b2, s2, t2, N, point0);

    // pillar scatter of point0 into the [B,512,512,32] grid, then f32 -> f16 NHWC
    k_zero4<<<(unsigned)((g32 / 4 + TB - 1) / TB), TB, 0, stream>>>(gridF, g32);
    k_scat<<<(unsigned)(((size_t)N * 32 + TB - 1) / TB), TB, 0, stream>>>(point0, gids, (unsigned*)gridF, N, 5);
    k_cvt<<<(unsigned)((g32 / 4 + TB - 1) / TB), TB, 0, stream>>>(gridF, gA, g32);

    // 4 dilated 3x3 WMMA convs, ping-pong gA <-> gB (final result lands in gA)
    const int dils[4] = {1, 2, 2, 2};
    _Float16* cin = gA;
    _Float16* cout = gB;
    for (int l = 0; l < 4; ++l) {
        k_conv<<<32768 / 8, TB, 0, stream>>>(cin, cout, wfrag + (size_t)l * 9 * 2 * 32 * 16,
                                             b2d + l * 32, s2d + l * 32, t2d + l * 32, dils[l]);
        _Float16* t = cin; cin = cout; cout = t;
    }

    // bilinear gather + final classifier
    k_final<<<(N + TB - 1) / TB, TB, 0, stream>>>(bids, pts, point0, cin, w3, b3, out, N);
}
